// GCNPathActor_50714973831671
// MI455X (gfx1250) — matrix operand order repaired and checked by reading, not compiled, not run on hardware
//
#include <hip/hip_runtime.h>
#include <hip/hip_bf16.h>

typedef __attribute__((ext_vector_type(16))) __bf16 v16bf;
typedef __attribute__((ext_vector_type(8)))  __bf16 v8bf;
typedef __attribute__((ext_vector_type(8)))  float  v8f;

#define HF 128   // hidden / feature width
#define LW 64    // path length

// ---------------- elementwise helpers ----------------

__global__ void fill_f32(float* p, float v, int n) {
  int i = blockIdx.x * blockDim.x + threadIdx.x;
  if (i < n) p[i] = v;
}

__global__ void degree_kernel(const long long* __restrict__ dst, float* __restrict__ deg, int E) {
  int e = blockIdx.x * blockDim.x + threadIdx.x;
  if (e < E) atomicAdd(&deg[(int)dst[e]], 1.0f);
}

__global__ void norm_kernel(float* nrm, int n) {
  int i = blockIdx.x * blockDim.x + threadIdx.x;
  if (i < n) nrm[i] = rsqrtf(fmaxf(nrm[i], 1.0f));
}

__global__ void cvt_bf16(const float* __restrict__ in, __bf16* __restrict__ out, long n) {
  long i = (long)blockIdx.x * blockDim.x + threadIdx.x;
  if (i < n) out[i] = (__bf16)in[i];
}

// in: K x N row-major f32  ->  out: N x K row-major bf16 (transposed)
__global__ void cvt_bf16_T(const float* __restrict__ in, __bf16* __restrict__ out, int K, int N) {
  int i = blockIdx.x * blockDim.x + threadIdx.x;
  if (i < K * N) {
    int k = i / N, c = i % N;
    out[c * K + k] = (__bf16)in[i];
  }
}

// ---------------- WMMA GEMM: C(MxN,f32) = A(MxK,bf16) * BT(NxK,bf16)^T ----------------
// K fixed at 128. Block = 256 threads (8 waves), processes RT=4 row tiles.
// BT staged into LDS once per block via global_load_async_to_lds_b128 (ASYNCcnt),
// B fragments held in registers across the 4 row tiles.
template <int N, bool BIAS, bool RELU>
__global__ __launch_bounds__(256) void wmma_gemm_bf16(
    const __bf16* __restrict__ A,
    const __bf16* __restrict__ BT,
    const float* __restrict__ bias,
    float* __restrict__ C,
    int Mtiles) {
  constexpr int RT = 4;          // row tiles per block
  constexpr int CT = N / 16 / 8; // col tiles per wave (1 for N=128, 2 for N=256)
  __shared__ __bf16 sB[N * 128];

  const int tid  = threadIdx.x;
  const int lane = tid & 31;
  const int wave = tid >> 5;
  const int lrow = lane & 15;
  const int half = lane >> 4;

  // ---- async stage BT (N x 128 bf16) into LDS ----
  {
    constexpr int CHUNKS = (N * 128 * 2) / 16;   // 16B chunks
    const char* gsrc = (const char*)BT;
    unsigned ldsbase = (unsigned)(uintptr_t)&sB[0];
#pragma unroll
    for (int c = tid; c < CHUNKS; c += 256) {
      unsigned dst = ldsbase + (unsigned)c * 16u;
      unsigned long long src = (unsigned long long)(uintptr_t)(gsrc + (size_t)c * 16);
      asm volatile("global_load_async_to_lds_b128 %0, %1, off"
                   :: "v"(dst), "v"(src) : "memory");
    }
    asm volatile("s_wait_asynccnt 0x0" ::: "memory");
  }
  __syncthreads();

  // ---- load this wave's B fragments from LDS into registers ----
  v16bf bfrag[CT][4];
#pragma unroll
  for (int ct = 0; ct < CT; ++ct) {
    const int c0 = (wave * CT + ct) * 16;
    const __bf16* bcol = &sB[(c0 + lrow) * 128];
#pragma unroll
    for (int kk = 0; kk < 4; ++kk)
      bfrag[ct][kk] = *(const v16bf*)(bcol + kk * 32 + 16 * half);
  }

  // ---- row tiles ----
  const int t0 = blockIdx.x * RT;
#pragma unroll
  for (int rt = 0; rt < RT; ++rt) {
    const int tile = t0 + rt;
    if (tile >= Mtiles) break;                   // wave-uniform
    const int r0 = tile * 16;
    const __bf16* arow = A + (size_t)(r0 + lrow) * 128;

    // A fragments: lanes 0-15 hold K [kk..kk+16); lanes 16-31 hold K [kk+16..kk+32)
    v16bf afrag[4];
#pragma unroll
    for (int kk = 0; kk < 4; ++kk) {
      const __bf16* ap0 = arow + kk * 32 + 16 * half;
      v8bf a0 = *(const v8bf*)(ap0);
      v8bf a1 = *(const v8bf*)(ap0 + 8);
      afrag[kk] = __builtin_shufflevector(a0, a1, 0,1,2,3,4,5,6,7,8,9,10,11,12,13,14,15);
    }

#pragma unroll
    for (int ct = 0; ct < CT; ++ct) {
      v8f acc = {};
#pragma unroll
      for (int kk = 0; kk < 4; ++kk)
        acc = __builtin_amdgcn_wmma_f32_16x16x32_bf16(
            false, afrag[kk], false, bfrag[ct][kk],
            (short)0, acc, false, false);
      const int col = (wave * CT + ct) * 16 + lrow;
      const float badd = BIAS ? bias[col] : 0.0f;
      float* crow = C + (size_t)(r0 + 8 * half) * N + col;
#pragma unroll
      for (int v = 0; v < 8; ++v) {
        float val = acc[v] + badd;
        if (RELU) val = fmaxf(val, 0.0f);
        crow[(size_t)v * N] = val;               // constant offsets: v*N*4 bytes
      }
    }
  }
}

// ---------------- edge scatter: agg[d] += coef * h[s], 32 lanes per edge ----------------
__global__ void edge_scatter(const long long* __restrict__ src,
                             const long long* __restrict__ dst,
                             const float* __restrict__ nrm,
                             const float* __restrict__ h,
                             float* __restrict__ agg, int E) {
  int tid = blockIdx.x * blockDim.x + threadIdx.x;
  int e = tid >> 5;
  if (e >= E) return;
  int s = (int)src[e], d = (int)dst[e];
  float coef = nrm[s] * nrm[d];
  int fo = (tid & 31) * 4;
  const float4 hv = *(const float4*)(h + (long)s * HF + fo);
  float* o = agg + (long)d * HF + fo;
  atomicAdd(o + 0, hv.x * coef);
  atomicAdd(o + 1, hv.y * coef);
  atomicAdd(o + 2, hv.z * coef);
  atomicAdd(o + 3, hv.w * coef);
}

// ---------------- finalize: self-loop + bias + relu (+ bf16 copy for next GEMM) ----------------
__global__ void finalize_layer(const float* __restrict__ agg,
                               const float* __restrict__ hh,
                               const float* __restrict__ nrm,
                               const float* __restrict__ bias,
                               float* __restrict__ out_f32,
                               __bf16* __restrict__ out_bf /*nullable*/) {
  int node = blockIdx.x, f = threadIdx.x;
  long i = (long)node * HF + f;
  float nv = nrm[node];
  float v = agg[i] + hh[i] * nv * nv + bias[f];
  v = fmaxf(v, 0.0f);
  out_f32[i] = v;
  if (out_bf) out_bf[i] = (__bf16)v;
}

// ---------------- ragged path mean-pool ----------------
__global__ void path_pool(const long long* __restrict__ paths,
                          const unsigned char* __restrict__ mask,
                          const float* __restrict__ h2,
                          float* __restrict__ pe, __bf16* __restrict__ peb) {
  int p = blockIdx.x, f = threadIdx.x;     // 128 threads
  float s = 0.0f, cnt = 0.0f;
  for (int l = 0; l < LW; ++l) {
    if (mask[p * LW + l]) {
      int node = (int)paths[p * LW + l];
      s += h2[(long)node * HF + f];
      cnt += 1.0f;
    }
  }
  float v = s / fmaxf(cnt, 1.0f);
  pe[(long)p * HF + f] = v;
  peb[(long)p * HF + f] = (__bf16)v;
}

// ---------------- final dot (Wm2) per path: one wave per path ----------------
__global__ void mlp2_scores(const float* __restrict__ mh,
                            const float* __restrict__ Wm2,
                            const float* __restrict__ bm2,
                            float* __restrict__ scores, int M, int P) {
  int p    = blockIdx.x * (blockDim.x >> 5) + (threadIdx.x >> 5);
  int lane = threadIdx.x & 31;
  if (p >= P) return;
  float s = 0.0f;
  for (int i = lane; i < M; i += 32) s += mh[(long)p * M + i] * Wm2[i];
  for (int off = 16; off; off >>= 1) s += __shfl_down(s, off, 32);
  if (lane == 0) scores[p] = s + bm2[0];
}

// ---------------- softmax over P (single block) ----------------
__global__ void softmax_block(const float* __restrict__ scores, float* __restrict__ out, int P) {
  __shared__ float red[1024];
  int t = threadIdx.x;
  float v = (t < P) ? scores[t] : -3.402823466e38f;
  red[t] = v; __syncthreads();
  for (int s = 512; s; s >>= 1) { if (t < s) red[t] = fmaxf(red[t], red[t + s]); __syncthreads(); }
  float mx = red[0]; __syncthreads();
  float e = (t < P) ? __expf(v - mx) : 0.0f;
  red[t] = e; __syncthreads();
  for (int s = 512; s; s >>= 1) { if (t < s) red[t] += red[t + s]; __syncthreads(); }
  if (t < P) out[t] = e / red[0];
}

// =====================================================================

extern "C" void kernel_launch(void* const* d_in, const int* in_sizes, int n_in,
                              void* d_out, int out_size, void* d_ws, size_t ws_size,
                              hipStream_t stream) {
  const float* x    = (const float*)d_in[0];
  const float* W1   = (const float*)d_in[1];
  const float* b1   = (const float*)d_in[2];
  const float* W2   = (const float*)d_in[3];
  const float* b2   = (const float*)d_in[4];
  const float* Wm1  = (const float*)d_in[5];
  const float* bm1  = (const float*)d_in[6];
  const float* Wm2  = (const float*)d_in[7];
  const float* bm2  = (const float*)d_in[8];
  const long long* eidx  = (const long long*)d_in[9];
  const long long* paths = (const long long*)d_in[10];
  const unsigned char* pmask = (const unsigned char*)d_in[11];

  const int Nn = in_sizes[0] / HF;        // 50000
  const int E  = in_sizes[9] / 2;         // 800000
  const int P  = in_sizes[10] / LW;       // 1024
  const int M  = 256;                     // MLP hidden
  const long long* esrc = eidx;
  const long long* edst = eidx + E;

  // ---- workspace layout ----
  char* base = (char*)d_ws;
  size_t off = 0;
  auto carve = [&](size_t bytes) -> char* {
    char* p = base + off;
    off = (off + bytes + 255) & ~(size_t)255;
    return p;
  };
  float*  nrm  = (float*)carve((size_t)Nn * 4);
  __bf16* xb   = (__bf16*)carve((size_t)Nn * HF * 2);
  __bf16* h1b  = (__bf16*)carve((size_t)Nn * HF * 2);
  __bf16* w1T  = (__bf16*)carve((size_t)HF * HF * 2);
  __bf16* w2T  = (__bf16*)carve((size_t)HF * HF * 2);
  __bf16* wm1T = (__bf16*)carve((size_t)HF * M * 2);
  float*  hh   = (float*)carve((size_t)Nn * HF * 4);   // GEMM output (pre-aggregation)
  float*  agg  = (float*)carve((size_t)Nn * HF * 4);   // scatter accumulator
  float*  hf   = (float*)carve((size_t)Nn * HF * 4);   // finalized layer output (h1 then h2)
  float*  pe   = (float*)carve((size_t)P * HF * 4);
  __bf16* peb  = (__bf16*)carve((size_t)P * HF * 2);
  float*  mh   = (float*)carve((size_t)P * M * 4);
  float*  sc   = (float*)carve((size_t)P * 4);
  (void)ws_size; (void)n_in; (void)out_size;

  // ---- degree -> norm ----
  fill_f32<<<(Nn + 255) / 256, 256, 0, stream>>>(nrm, 1.0f, Nn);          // self loop
  degree_kernel<<<(E + 255) / 256, 256, 0, stream>>>(edst, nrm, E);
  norm_kernel<<<(Nn + 255) / 256, 256, 0, stream>>>(nrm, Nn);

  // ---- bf16 conversions ----
  long nx = (long)Nn * HF;
  cvt_bf16<<<(int)((nx + 255) / 256), 256, 0, stream>>>(x, xb, nx);
  cvt_bf16_T<<<(HF * HF + 255) / 256, 256, 0, stream>>>(W1, w1T, HF, HF);
  cvt_bf16_T<<<(HF * HF + 255) / 256, 256, 0, stream>>>(W2, w2T, HF, HF);
  cvt_bf16_T<<<(HF * M + 255) / 256, 256, 0, stream>>>(Wm1, wm1T, HF, M);

  const int mtiles = Nn / 16;                           // 3125
  const int gemm_blocks = (mtiles + 3) / 4;             // 4 row tiles per block
  const int escatter_blocks = (E * 32 + 255) / 256;

  // ---- layer 1 ----
  wmma_gemm_bf16<128, false, false><<<gemm_blocks, 256, 0, stream>>>(xb, w1T, nullptr, hh, mtiles);
  hipMemsetAsync(agg, 0, (size_t)Nn * HF * 4, stream);
  edge_scatter<<<escatter_blocks, 256, 0, stream>>>(esrc, edst, nrm, hh, agg, E);
  finalize_layer<<<Nn, HF, 0, stream>>>(agg, hh, nrm, b1, hf, h1b);

  // ---- layer 2 ----
  wmma_gemm_bf16<128, false, false><<<gemm_blocks, 256, 0, stream>>>(h1b, w2T, nullptr, hh, mtiles);
  hipMemsetAsync(agg, 0, (size_t)Nn * HF * 4, stream);
  edge_scatter<<<escatter_blocks, 256, 0, stream>>>(esrc, edst, nrm, hh, agg, E);
  finalize_layer<<<Nn, HF, 0, stream>>>(agg, hh, nrm, b2, hf, nullptr);

  // ---- path pooling ----
  path_pool<<<P, HF, 0, stream>>>(paths, pmask, hf, pe, peb);

  // ---- MLP: (P x 128) @ (128 x 256), bias + relu fused ----
  wmma_gemm_bf16<256, true, true><<<(P / 16 + 3) / 4, 256, 0, stream>>>(peb, wm1T, bm1, mh, P / 16);

  // ---- final dot + softmax ----
  mlp2_scores<<<(P + 7) / 8, 256, 0, stream>>>(mh, Wm2, bm2, sc, M, P);
  softmax_block<<<1, 1024, 0, stream>>>(sc, (float*)d_out, P);
}
